// BitwiseLinear_6871947674178
// MI455X (gfx1250) — 8-bit matrix operand order repaired and checked by reading, compiled, not run on hardware
//
#include <hip/hip_runtime.h>
#include <hip/hip_bf16.h>

#define NB   32768
#define NK   1024
#define NO   1024

typedef int v8i __attribute__((ext_vector_type(8)));
typedef int v4i __attribute__((ext_vector_type(4)));
typedef int v2i __attribute__((ext_vector_type(2)));

__device__ __forceinline__ int sgn8(float v) { return (v > 0.0f) - (v < 0.0f); }

__global__ __launch_bounds__(256) void quant_x_kernel(
    const float* __restrict__ x, signed char* __restrict__ a8,
    float* __restrict__ in_scale)
{
    const int lane = threadIdx.x & 31;
    const int wave = threadIdx.x >> 5;
    const int row  = blockIdx.x * 8 + wave;

    const float*  px = x  + (size_t)row * NK;
    signed char*  pq = a8 + (size_t)row * NK;

    float asum = 0.0f;
#pragma unroll
    for (int i = 0; i < 8; ++i) {
        const int c = i * 128 + lane * 4;
        float4 v = *(const float4*)(px + c);
        asum += fabsf(v.x) + fabsf(v.y) + fabsf(v.z) + fabsf(v.w);
        unsigned int packed = ((unsigned)(sgn8(v.x) & 0xFF))
                            | ((unsigned)(sgn8(v.y) & 0xFF) << 8)
                            | ((unsigned)(sgn8(v.z) & 0xFF) << 16)
                            | ((unsigned)(sgn8(v.w) & 0xFF) << 24);
        *(unsigned int*)(pq + c) = packed;
    }
#pragma unroll
    for (int off = 16; off > 0; off >>= 1)
        asum += __shfl_xor(asum, off, 32);
    if (lane == 0) in_scale[row] = asum * (1.0f / (float)NK);
}

__global__ __launch_bounds__(256) void quant_w_kernel(
    const float* __restrict__ w, const float* __restrict__ g,
    signed char* __restrict__ w8, float* __restrict__ w_scale)
{
    const int lane = threadIdx.x & 31;
    const int wave = threadIdx.x >> 5;
    const int row  = blockIdx.x * 8 + wave;

    const float*  pw = w  + (size_t)row * NK;
    const float*  pg = g  + (size_t)row * NK;
    signed char*  pq = w8 + (size_t)row * NK;

    float asum = 0.0f;
#pragma unroll
    for (int i = 0; i < 8; ++i) {
        const int c = i * 128 + lane * 4;
        float4 vw = *(const float4*)(pw + c);
        float4 vg = *(const float4*)(pg + c);
        asum += fabsf(vw.x) + fabsf(vw.y) + fabsf(vw.z) + fabsf(vw.w);
        int b0 = (vg.x < 0.0f) ? 0 : sgn8(vw.x);
        int b1 = (vg.y < 0.0f) ? 0 : sgn8(vw.y);
        int b2 = (vg.z < 0.0f) ? 0 : sgn8(vw.z);
        int b3 = (vg.w < 0.0f) ? 0 : sgn8(vw.w);
        unsigned int packed = ((unsigned)(b0 & 0xFF))
                            | ((unsigned)(b1 & 0xFF) << 8)
                            | ((unsigned)(b2 & 0xFF) << 16)
                            | ((unsigned)(b3 & 0xFF) << 24);
        *(unsigned int*)(pq + c) = packed;
    }
#pragma unroll
    for (int off = 16; off > 0; off >>= 1)
        asum += __shfl_xor(asum, off, 32);
    if (lane == 0) w_scale[row] = asum * (1.0f / (float)NK);
}

__global__ __launch_bounds__(256) void gemm_iu8_kernel(
    const signed char* __restrict__ a8, const signed char* __restrict__ w8,
    const float* __restrict__ in_scale, const float* __restrict__ w_scale,
    const float* __restrict__ bias, float* __restrict__ out)
{
    const int lane = threadIdx.x;
    const int m0   = blockIdx.x * 64;
    const int n0   = (blockIdx.y * 8 + threadIdx.y) * 16;
    const int h    = lane >> 4;
    const int l15  = lane & 15;

    const signed char* pb = w8 + (size_t)(n0 + l15) * NK + 16 * h;
    const signed char* pa = a8 + (size_t)(m0 + l15) * NK + 16 * h;

    v8i acc[4] = {};

    for (int k0 = 0; k0 < NK; k0 += 64) {
        v4i b03 = *(const v4i*)(pb + k0);
        v4i b47 = *(const v4i*)(pb + k0 + 32);
        v8i bf = { b03[0], b03[1], b03[2], b03[3],
                   b47[0], b47[1], b47[2], b47[3] };
#pragma unroll
        for (int s = 0; s < 4; ++s) {
            const signed char* pas = pa + (size_t)(16 * s) * NK + k0;
            v2i a0 = *(const v2i*)(pas);
            v2i a1 = *(const v2i*)(pas + 8);
            v2i a2 = *(const v2i*)(pas + 32);
            v2i a3 = *(const v2i*)(pas + 40);
            v8i af = { a0[0], a0[1], a1[0], a1[1],
                       a2[0], a2[1], a3[0], a3[1] };
            acc[s] = __builtin_amdgcn_wmma_i32_16x16x64_iu8(
                true, af, true, bf, acc[s], false, false);
        }
    }

    const float wsc = w_scale[n0 + l15];
    const float bs  = bias[n0 + l15];
#pragma unroll
    for (int s = 0; s < 4; ++s) {
        const int   rbase = m0 + 16 * s + 8 * h;
        float*      po    = out + (size_t)rbase * NO + n0 + l15;
#pragma unroll
        for (int r = 0; r < 8; ++r) {
            const float sc = in_scale[rbase + r];
            po[(size_t)r * NO] = sc * wsc * ((float)acc[s][r] + bs);
        }
    }
}

extern "C" void kernel_launch(void* const* d_in, const int* in_sizes, int n_in,
                              void* d_out, int out_size, void* d_ws, size_t ws_size,
                              hipStream_t stream) {
    (void)in_sizes; (void)n_in; (void)out_size; (void)ws_size;
    const float* x      = (const float*)d_in[0];
    const float* weight = (const float*)d_in[1];
    const float* gate   = (const float*)d_in[2];
    const float* bias   = (const float*)d_in[3];
    float*       out    = (float*)d_out;

    signed char* a8       = (signed char*)d_ws;
    signed char* w8       = a8 + (size_t)NB * NK;
    float*       in_scale = (float*)(w8 + (size_t)NO * NK);
    float*       w_scale  = in_scale + NB;

    quant_x_kernel<<<dim3(NB / 8), dim3(256), 0, stream>>>(x, a8, in_scale);
    quant_w_kernel<<<dim3(NO / 8), dim3(256), 0, stream>>>(weight, gate, w8, w_scale);
    gemm_iu8_kernel<<<dim3(NB / 64, NO / 128), dim3(32, 8), 0, stream>>>(
        a8, w8, in_scale, w_scale, bias, out);
}
